// CBAM_80178449482542
// MI455X (gfx1250) — matrix operand order repaired and checked by reading, not compiled, not run on hardware
//
#include <hip/hip_runtime.h>
#include <math.h>

#define INC   64
#define BATCH 8
#define HID   16

typedef __attribute__((ext_vector_type(16))) _Float16 v16h;
typedef __attribute__((ext_vector_type(8)))  float    v8f;

#define ENC_NEG_INF 0x007FFFFFu   // order-preserving encoding of -inf

__device__ __forceinline__ unsigned f2ord(float f) {
  unsigned u = __float_as_uint(f);
  return (u & 0x80000000u) ? ~u : (u | 0x80000000u);
}
__device__ __forceinline__ float ord2f(unsigned u) {
  return (u & 0x80000000u) ? __uint_as_float(u & 0x7FFFFFFFu)
                           : __uint_as_float(~u);
}

// ---------------------------------------------------------------- K0: init
__global__ void k_init(float* g_cnt, float* g_sum, unsigned* g_maxe,
                       float* sm_pad) {
  int t = blockIdx.x * blockDim.x + threadIdx.x;
  if (t < BATCH * INC) { g_sum[t] = 0.f; g_maxe[t] = ENC_NEG_INF; }
  if (t < BATCH) g_cnt[t] = 0.f;
  if (t < 2) sm_pad[t] = 0.f;   // zero pad row (index N) for neighbor gather
}

// ------------------------------------------------- K1: segmented sum / max
__global__ __launch_bounds__(256)
void k_reduce(const float* __restrict__ F, const int* __restrict__ bidx,
              float* __restrict__ g_sum, unsigned* __restrict__ g_maxe,
              float* __restrict__ g_cnt, int n) {
  __shared__ float    s_sum[BATCH * INC];
  __shared__ unsigned s_max[BATCH * INC];
  __shared__ float    s_cnt[BATCH];
  int t = threadIdx.x;
  for (int i = t; i < BATCH * INC; i += 256) { s_sum[i] = 0.f; s_max[i] = ENC_NEG_INF; }
  if (t < BATCH) s_cnt[t] = 0.f;
  __syncthreads();

  int c = t & 63;          // channel
  int r = t >> 6;          // 0..3 rows per block-pass
  for (int p = blockIdx.x * 4 + r; p < n; p += gridDim.x * 4) {
    int b = bidx[p];
    float v = F[(size_t)p * INC + c];
    atomicAdd(&s_sum[b * INC + c], v);
    atomicMax(&s_max[b * INC + c], f2ord(v));
    if (c == 0) atomicAdd(&s_cnt[b], 1.f);
  }
  __syncthreads();
  for (int i = t; i < BATCH * INC; i += 256) {
    atomicAdd(&g_sum[i], s_sum[i]);
    atomicMax(&g_maxe[i], s_max[i]);
  }
  if (t < BATCH) atomicAdd(&g_cnt[t], s_cnt[t]);
}

// ------------------------------------------- K2: per-batch MLP gate (WMMA)
// rows 0-7 of X = seg_mean, rows 8-15 = seg_max; gate[b] = sigmoid(out[b]+out[b+8])
__global__ __launch_bounds__(32)
void k_gate(const float* __restrict__ g_sum, const unsigned* __restrict__ g_maxe,
            const float* __restrict__ g_cnt,
            const float* __restrict__ W1, const float* __restrict__ b1,
            const float* __restrict__ W2, const float* __restrict__ b2,
            float* __restrict__ gate) {
  __shared__ _Float16 Xh[16][64];    // input rows (f16)
  __shared__ _Float16 W1h[64][16];
  __shared__ _Float16 Hp[16][32];    // hidden, K-padded to 32 (upper half zero)
  __shared__ _Float16 W2h[32][64];   // rows 16-31 zero
  __shared__ float b1s[16];
  __shared__ float b2s[64];
  int t = threadIdx.x;

  for (int i = t; i < BATCH * INC; i += 32) {
    int bb = i >> 6, c = i & 63;
    Xh[bb][c]     = (_Float16)(g_sum[i] / g_cnt[bb]);
    Xh[bb + 8][c] = (_Float16)ord2f(g_maxe[i]);
  }
  for (int i = t; i < 64 * 16; i += 32) W1h[i >> 4][i & 15] = (_Float16)W1[i];
  for (int i = t; i < 32 * 64; i += 32) {
    int rr = i >> 6, c = i & 63;
    W2h[rr][c] = (rr < HID) ? (_Float16)W2[i] : (_Float16)0.f;
  }
  if (t < 16) b1s[t] = b1[t];
  for (int i = t; i < 64; i += 32) b2s[i] = b2[i];
  for (int i = t; i < 256; i += 32) Hp[i >> 4][16 + (i & 15)] = (_Float16)0.f;
  __syncthreads();

  int m  = t & 15;    // row (A) / col (B) within tile
  int kg = t >> 4;    // lane-half selects K-subgroup

  // ---- layer 1: [16,64] @ [64,16], K = 2 x 32 ----
  v8f acc1 = {};
  for (int kt = 0; kt < 2; ++kt) {
    v16h af, bf;
#pragma unroll
    for (int j = 0; j < 8; ++j) {
      af[j]     = Xh[m][kt * 32 + kg * 8 + j];
      af[8 + j] = Xh[m][kt * 32 + 16 + kg * 8 + j];
    }
#pragma unroll
    for (int i = 0; i < 16; ++i) bf[i] = W1h[kt * 32 + (i >> 3) * 16 + kg * 8 + (i & 7)][m];
    acc1 = __builtin_amdgcn_wmma_f32_16x16x32_f16(false, af, false, bf,
                                                  (short)0, acc1, false, false);
  }
  // bias + relu -> Hp  (acc1[r]: lanes<16 -> M=r, lanes>=16 -> M=r+8; N=m)
#pragma unroll
  for (int r = 0; r < 8; ++r) {
    float v = acc1[r] + b1s[m];
    Hp[r + kg * 8][m] = (_Float16)fmaxf(v, 0.f);
  }
  __syncthreads();

  // ---- layer 2: [16,16(pad 32)] @ [16(pad 32),64], 4 N-tiles ----
  v16h a2;
#pragma unroll
  for (int j = 0; j < 8; ++j) {
    a2[j]     = Hp[m][kg * 8 + j];
    a2[8 + j] = Hp[m][16 + kg * 8 + j];   // zeros
  }
  for (int nt = 0; nt < 4; ++nt) {
    v16h b2f;
#pragma unroll
    for (int i = 0; i < 16; ++i) b2f[i] = W2h[(i >> 3) * 16 + kg * 8 + (i & 7)][nt * 16 + m];
    v8f acc2 = {};
    acc2 = __builtin_amdgcn_wmma_f32_16x16x32_f16(false, a2, false, b2f,
                                                  (short)0, acc2, false, false);
#pragma unroll
    for (int r = 0; r < 8; ++r) {
      float v  = acc2[r];
      float vo = __shfl_xor(v, 16, 32);   // pair mean-row (M=r) with max-row (M=r+8)
      if (kg == 0) {
        int ch = nt * 16 + m;
        float s = v + vo + 2.f * b2s[ch]; // b2 added in both mlp() calls
        gate[r * INC + ch] = 1.f / (1.f + __expf(-s));
      }
    }
  }
}

// ------------------------------- K3: z = F*gate, per-point mean/max -> sm
__global__ __launch_bounds__(256)
void k_zsm(const float* __restrict__ F, const int* __restrict__ bidx,
           const float* __restrict__ gate, float2* __restrict__ sm, int n) {
  int t = threadIdx.x;
  int lane = t & 31, wid = t >> 5;
  int l16 = lane & 15;
  int p = (blockIdx.x * 8 + wid) * 2 + (lane >> 4);   // 16 lanes per point
  float sum = 0.f, mx = -INFINITY;
  if (p < n) {
    int b = bidx[p];
    const float4* F4 = (const float4*)F;
    const float4* G4 = (const float4*)gate;
    float4 f = F4[(size_t)p * 16 + l16];
    float4 g = G4[b * 16 + l16];
    float z0 = f.x * g.x, z1 = f.y * g.y, z2 = f.z * g.z, z3 = f.w * g.w;
    sum = z0 + z1 + z2 + z3;
    mx  = fmaxf(fmaxf(z0, z1), fmaxf(z2, z3));
  }
#pragma unroll
  for (int msk = 1; msk < 16; msk <<= 1) {
    sum += __shfl_xor(sum, msk, 32);
    mx   = fmaxf(mx, __shfl_xor(mx, msk, 32));
  }
  if (p < n && l16 == 0) sm[p] = make_float2(sum * (1.f / 64.f), mx);
}

// ------------------- K4: 27-pt sparse conv -> sigmoid -> out = z * sig(c)
__global__ __launch_bounds__(256)
void k_spatial(const float* __restrict__ F, const int* __restrict__ bidx,
               const int* __restrict__ nbr, const float2* __restrict__ sm,
               const float* __restrict__ gate, const float* __restrict__ conv_w,
               float* __restrict__ out, int n) {
  __shared__ float cw0[27], cw1[27];
  __shared__ float sg[BATCH * INC];
  int t = threadIdx.x;
  if (t < 27) { cw0[t] = conv_w[t * 2]; cw1[t] = conv_w[t * 2 + 1]; }
  for (int i = t; i < BATCH * INC; i += 256) sg[i] = gate[i];
  __syncthreads();

  int lane = t & 31, wid = t >> 5;
  int p = blockIdx.x * 8 + wid;                       // one wave per point
  if (p >= n) return;

  float acc = 0.f;
  if (lane < 27) {
    int j = nbr[(size_t)p * 27 + lane];               // j == n -> zero pad row
    float2 s = sm[j];
    acc = s.x * cw0[lane] + s.y * cw1[lane];
  }
#pragma unroll
  for (int msk = 1; msk < 32; msk <<= 1) acc += __shfl_xor(acc, msk, 32);
  float sig = 1.f / (1.f + __expf(-acc));

  int b = bidx[p];
  const float2* F2 = (const float2*)F;
  const float2* G2 = (const float2*)sg;
  float2 f = F2[(size_t)p * 32 + lane];
  float2 g = G2[b * 32 + lane];
  float2 o; o.x = f.x * g.x * sig; o.y = f.y * g.y * sig;
  ((float2*)out)[(size_t)p * 32 + lane] = o;
}

// -------------------------------------------------------------- launcher
extern "C" void kernel_launch(void* const* d_in, const int* in_sizes, int n_in,
                              void* d_out, int out_size, void* d_ws, size_t ws_size,
                              hipStream_t stream) {
  const float* F    = (const float*)d_in[0];
  const int*   bidx = (const int*)d_in[1];
  const int*   nbr  = (const int*)d_in[2];
  const float* W1   = (const float*)d_in[3];
  const float* b1   = (const float*)d_in[4];
  const float* W2   = (const float*)d_in[5];
  const float* b2   = (const float*)d_in[6];
  const float* cw   = (const float*)d_in[7];
  int n = in_sizes[0] / INC;

  float*    ws     = (float*)d_ws;
  float*    g_cnt  = ws;                            // 8
  float*    g_sum  = ws + 8;                        // 512
  unsigned* g_maxe = (unsigned*)(ws + 8 + 512);     // 512
  float*    gate   = ws + 8 + 512 + 512;            // 512
  float2*   sm     = (float2*)(ws + 1544);          // (n+1) x float2, 8B aligned

  k_init   <<<2, 256, 0, stream>>>(g_cnt, g_sum, g_maxe, (float*)(sm + n));
  k_reduce <<<1024, 256, 0, stream>>>(F, bidx, g_sum, g_maxe, g_cnt, n);
  k_gate   <<<1, 32, 0, stream>>>(g_sum, g_maxe, g_cnt, W1, b1, W2, b2, gate);
  k_zsm    <<<(n + 15) / 16, 256, 0, stream>>>(F, bidx, gate, sm, n);
  k_spatial<<<(n + 7) / 8, 256, 0, stream>>>(F, bidx, nbr, sm, gate, cw,
                                             (float*)d_out, n);
}
